// InstructionsModel_88725434401326
// MI455X (gfx1250) — matrix operand order repaired and checked by reading, not compiled, not run on hardware
//
#include <hip/hip_runtime.h>
#include <hip/hip_bf16.h>

// ---- model dims ----
#define B_      32
#define S_      128
#define D_      300
#define DP      320      // D padded to 10 K-tiles of 32
#define VOCAB_  20000
#define VPAD    20032    // 313 chunks of 64
#define KTILES  10
#define CH      64       // vocab rows per chunk
#define NCHUNK  313
#define NROWS   4096     // B_*S_
#define CHBYTES (CH * DP * 2)   // 40960 B per chunk, contiguous in vocab_b

typedef __attribute__((ext_vector_type(16))) __bf16 v16bf;
typedef __attribute__((ext_vector_type(8)))  float  v8f;

struct BF16x16 { uint4 lo, hi; };

// 2 x 8 bf16 at p and p+16 (16B-aligned), table K order
__device__ __forceinline__ v16bf load16bf_contig(const __bf16* p) {
    BF16x16 t;
    t.lo = *(const uint4*)(p);
    t.hi = *(const uint4*)(p + 16);
    return __builtin_bit_cast(v16bf, t);
}

// A-matrix fragment (16-bit A 16x32 layout, ISA 7.12.2):
// lane<16: K = 0..7, 16..23 ; lane>=16: K = 8..15, 24..31.
__device__ __forceinline__ v16bf loadAfrag(const __bf16* rowp, int ktbase, int half8) {
    BF16x16 t;
    t.lo = *(const uint4*)(rowp + ktbase + half8);
    t.hi = *(const uint4*)(rowp + ktbase + half8 + 16);
    return __builtin_bit_cast(v16bf, t);
}

// generic -> LDS 32-bit offset for inline asm operands
__device__ __forceinline__ unsigned lds_off(const void* p) {
    return (unsigned)(uintptr_t)(const __attribute__((address_space(3))) void*)p;
}

// ---------------- prep kernels ----------------

// row-major bf16 vocab [VPAD][DP] (score-side B operand; chunk = contiguous 40KB)
__global__ void convert_vocab_kernel(const float* __restrict__ vocab,
                                     __bf16* __restrict__ vb) {
    int idx = blockIdx.x * 256 + threadIdx.x;
    if (idx >= VPAD * DP) return;
    int r = idx / DP, c = idx % DP;
    float v = (r < VOCAB_ && c < D_) ? vocab[r * D_ + c] : 0.0f;
    vb[idx] = (__bf16)v;
}

// transposed bf16 vocab [DP][VPAD] (value-side B operand; coalesced writes)
__global__ void convert_vocabT_kernel(const float* __restrict__ vocab,
                                      __bf16* __restrict__ vbt) {
    int idx = blockIdx.x * 256 + threadIdx.x;
    if (idx >= DP * VPAD) return;
    int c = idx / VPAD, r = idx % VPAD;
    float v = (r < VOCAB_ && c < D_) ? vocab[r * D_ + c] : 0.0f;
    vbt[idx] = (__bf16)v;
}

// X = words @ W   (4096 x 320, padded cols zero). f32 + bf16 copies.
__global__ void x_gemm_kernel(const float* __restrict__ words,
                              const float* __restrict__ W,
                              float* __restrict__ Xf,
                              __bf16* __restrict__ Xb) {
    int idx = blockIdx.x * 256 + threadIdx.x;
    if (idx >= NROWS * DP) return;
    int r = idx / DP, e = idx % DP;
    float a = 0.0f;
    if (e < D_) {
        const float* wr = &words[r * D_];
        for (int d = 0; d < D_; ++d) a += wr[d] * W[d * D_ + e];
    }
    Xf[idx] = a;
    Xb[idx] = (__bf16)a;
}

// logit of the "default embed" column: ldef[r] = X[r,:] . default_embed
__global__ void ldef_kernel(const float* __restrict__ Xf,
                            const float* __restrict__ de,
                            float* __restrict__ ldef) {
    int r = blockIdx.x * 256 + threadIdx.x;
    if (r >= NROWS) return;
    float a = 0.0f;
    for (int d = 0; d < D_; ++d) a += Xf[r * DP + d] * de[d];
    ldef[r] = a;
}

// ---------------- fused vocab softmax-mixture (WMMA + async LDS) -------------
// Per workgroup: 16 token rows, 4 waves. Flash-style online softmax over
// 20001 logits fused with V = P[:, :-1] @ vocab + P[:, -1] * words.
// The next vocab chunk is prefetched global->LDS with async b128 copies
// (ASYNCcnt), hiding L2 latency with zero VGPR cost.
__global__ void __launch_bounds__(128)
vocab_attend_kernel(const __bf16* __restrict__ vocab_b,   // [VPAD][DP]
                    const __bf16* __restrict__ vocabT_b,  // [DP][VPAD]
                    const __bf16* __restrict__ Xb,        // [NROWS][DP]
                    const float*  __restrict__ ldef,      // [NROWS]
                    const float*  __restrict__ words,     // [NROWS][D_]
                    float* __restrict__ Vf)               // [NROWS][DP]
{
    __shared__ __align__(16) __bf16 vchunk[CH * DP];   // 40KB vocab chunk (flat)
    __shared__ __align__(16) __bf16 pbuf[16][72];      // P tile (16 x 64), padded stride
    __shared__ float sm[16], sl[16];                   // running max / denom
    __shared__ float rowmax[4][16], rowsum[4][16];

    const int tid  = threadIdx.x;
    const int w    = tid >> 5;          // wave id 0..3
    const int lane = tid & 31;
    const int half = (lane >= 16) ? 1 : 0;
    const int n    = lane & 15;
    const int row0 = blockIdx.x * 16;

    if (tid < 16) { sm[tid] = ldef[row0 + tid]; sl[tid] = 1.0f; }  // default col seeds stats

    // persistent A fragments of X (10 K-tiles), straight from global
    v16bf xa[KTILES];
    {
        const __bf16* xrow = Xb + (size_t)(row0 + n) * DP;
#pragma unroll
        for (int kt = 0; kt < KTILES; ++kt)
            xa[kt] = loadAfrag(xrow, kt * 32, half * 8);
    }

    // per-wave accumulator: 5 N-tiles (80 output columns)
    v8f acc[5];
#pragma unroll
    for (int t = 0; t < 5; ++t) acc[t] = (v8f){0,0,0,0,0,0,0,0};
    const int ncol0 = w * 80;

    // async copy of chunk 0 into LDS (flat 40KB: 128 thr x 20 x 16B)
    const unsigned lbase = lds_off(&vchunk[0]) + tid * 16;
    {
        const int g0 = tid * 16;
#pragma unroll
        for (int j = 0; j < 20; ++j) {
            asm volatile("global_load_async_to_lds_b128 %0, %1, %2 offset:0"
                         :: "v"(lbase + j * 2048), "v"(g0 + j * 2048), "s"(vocab_b)
                         : "memory");
        }
    }

    for (int c = 0; c < NCHUNK; ++c) {
        const int v0 = c * CH;

        // chunk resident in LDS: wait own wave's async copies, then all waves
        asm volatile("s_wait_asynccnt 0x0" ::: "memory");
        __syncthreads();

        // ---- scores: S(16x16) = X(16x320) @ vocab_tileT, 10 WMMAs ----
        // B from the LDS chunk (clustered ds_load_b128); two accumulators
        // break the WMMA dependency chain.
        v8f S0 = (v8f){0,0,0,0,0,0,0,0};
        v8f S1 = (v8f){0,0,0,0,0,0,0,0};
        const __bf16* vrow = &vchunk[(w * 16 + n) * DP + half * 8];
#pragma unroll
        for (int kt = 0; kt < KTILES; kt += 2) {
            v16bf b0 = load16bf_contig(vrow + kt * 32);
            v16bf b1 = load16bf_contig(vrow + (kt + 1) * 32);
            S0 = __builtin_amdgcn_wmma_f32_16x16x32_bf16(false, xa[kt],     false, b0,
                                                         (short)0, S0, false, false);
            S1 = __builtin_amdgcn_wmma_f32_16x16x32_bf16(false, xa[kt + 1], false, b1,
                                                         (short)0, S1, false, false);
        }

        // ---- mask tail rows; per-row max across the 16 lanes of the half ----
        const int v = v0 + w * 16 + n;
        float sc[8];
#pragma unroll
        for (int j = 0; j < 8; ++j) {
            sc[j] = (v < VOCAB_) ? (S0[j] + S1[j]) : -1e30f;
            float rm = sc[j];
#pragma unroll
            for (int off = 1; off < 16; off <<= 1)
                rm = fmaxf(rm, __shfl_xor(rm, off, 32));
            if (n == 0) rowmax[w][half * 8 + j] = rm;
        }
        __syncthreads();   // barrier 1: rowmax visible; all chunk reads retired
                           // (every score ds load is consumed by a WMMA above)

        // ---- issue async copy of the NEXT chunk into the same buffer ----
        // Safe: nothing below reads vchunk; completion fenced at next loop top.
        {
            const int cn = (c + 1 < NCHUNK) ? (c + 1) : 0;   // wrap: harmless reload
            const int g0 = cn * CHBYTES + tid * 16;
#pragma unroll
            for (int j = 0; j < 20; ++j) {
                asm volatile("global_load_async_to_lds_b128 %0, %1, %2 offset:0"
                             :: "v"(lbase + j * 2048), "v"(g0 + j * 2048), "s"(vocab_b)
                             : "memory");
            }
        }

        // ---- online softmax update ----
        float alpha[8], mnew[8];
#pragma unroll
        for (int j = 0; j < 8; ++j) {
            const int r = half * 8 + j;
            float mold = sm[r];
            float mn = mold;
            mn = fmaxf(mn, rowmax[0][r]); mn = fmaxf(mn, rowmax[1][r]);
            mn = fmaxf(mn, rowmax[2][r]); mn = fmaxf(mn, rowmax[3][r]);
            mnew[j]  = mn;
            alpha[j] = __expf(mold - mn);
            float p  = __expf(sc[j] - mn);    // masked -> 0
            float rs = p;
#pragma unroll
            for (int off = 1; off < 16; off <<= 1)
                rs += __shfl_xor(rs, off, 32);
            if (n == 0) rowsum[w][r] = rs;
            pbuf[r][w * 16 + n] = (__bf16)p;
#pragma unroll
            for (int t = 0; t < 5; ++t) acc[t][j] *= alpha[j];
        }
        __syncthreads();   // barrier 2: pbuf + rowsum visible

        if (w == 0 && n == 0) {   // lanes 0 & 16 update all 16 rows
#pragma unroll
            for (int j = 0; j < 8; ++j) {
                const int r = half * 8 + j;
                sl[r] = sl[r] * alpha[j]
                      + rowsum[0][r] + rowsum[1][r] + rowsum[2][r] + rowsum[3][r];
                sm[r] = mnew[j];
            }
        }

        // ---- value accumulate: acc(16x80) += P(16x64) @ vocab_chunk(64x80) ----
        v16bf pa0 = loadAfrag(&pbuf[n][0], 0,  half * 8);   // ds b128 x2
        v16bf pa1 = loadAfrag(&pbuf[n][0], 32, half * 8);
#pragma unroll
        for (int t = 0; t < 5; ++t) {
            const int ncol = ncol0 + t * 16 + n;
            const __bf16* vb = vocabT_b + (size_t)ncol * VPAD + v0 + half * 8;
            v16bf b0 = load16bf_contig(vb);        // K = 0..31 slice for this lane
            v16bf b1 = load16bf_contig(vb + 32);   // K = 32..63 slice
            acc[t] = __builtin_amdgcn_wmma_f32_16x16x32_bf16(false, pa0, false, b0,
                                                             (short)0, acc[t], false, false);
            acc[t] = __builtin_amdgcn_wmma_f32_16x16x32_bf16(false, pa1, false, b1,
                                                             (short)0, acc[t], false, false);
        }
    }

    // ---- finalize: V = (acc + exp(ldef - m) * words_row) / l ----
#pragma unroll
    for (int j = 0; j < 8; ++j) {
        const int r = half * 8 + j;
        const int grow = row0 + r;
        const float dscale = __expf(ldef[grow] - sm[r]);
        const float invl = 1.0f / sl[r];
#pragma unroll
        for (int t = 0; t < 5; ++t) {
            const int col = ncol0 + t * 16 + n;
            float out = 0.0f;
            if (col < D_)
                out = (acc[t][j] + dscale * words[grow * D_ + col]) * invl;
            Vf[(size_t)grow * DP + col] = out;
        }
    }
}

// ---------------- LSTM (batch-parallel, f32) ----------------
__global__ void __launch_bounds__(256)
lstm_kernel(const float* __restrict__ Vf, const int* __restrict__ lengths,
            const float* __restrict__ Wih, const float* __restrict__ Whh,
            const float* __restrict__ bih, const float* __restrict__ bhh,
            float* __restrict__ q) {
    __shared__ float h[D_], cst[D_], gbuf[4 * D_];
    const int b = blockIdx.x, tid = threadIdx.x;
    for (int i = tid; i < D_; i += 256) { h[i] = 0.0f; cst[i] = 0.0f; }
    __syncthreads();
    const int len = lengths[b];       // valid mask is a prefix -> just stop early
    for (int t = 0; t < len; ++t) {
        const float* x = &Vf[(size_t)(b * S_ + t) * DP];
        for (int g = tid; g < 4 * D_; g += 256) {
            float a = bih[g] + bhh[g];
            const float* wi = &Wih[g * D_];
            const float* wh = &Whh[g * D_];
            for (int k = 0; k < D_; ++k) a += x[k] * wi[k] + h[k] * wh[k];
            gbuf[g] = a;
        }
        __syncthreads();
        for (int d = tid; d < D_; d += 256) {
            float ig = gbuf[d], fg = gbuf[D_ + d], gg = gbuf[2 * D_ + d], og = gbuf[3 * D_ + d];
            float si = 1.0f / (1.0f + __expf(-ig));
            float sf = 1.0f / (1.0f + __expf(-fg));
            float so = 1.0f / (1.0f + __expf(-og));
            float cn = sf * cst[d] + si * tanhf(gg);
            cst[d] = cn;
            h[d] = so * tanhf(cn);
        }
        __syncthreads();
    }
    for (int i = tid; i < D_; i += 256) q[b * D_ + i] = h[i];
}

// ---------------- RNN readout (q fixed -> precompute u) ----------------
__global__ void __launch_bounds__(256)
rnn_kernel(const float* __restrict__ q,
           const float* __restrict__ Wih, const float* __restrict__ Whh,
           const float* __restrict__ bih, const float* __restrict__ bhh,
           float* __restrict__ H) {
    __shared__ float u[D_], h[D_], hn[D_];
    const int b = blockIdx.x, tid = threadIdx.x;
    for (int d = tid; d < D_; d += 256) {
        float a = bih[d] + bhh[d];
        const float* wi = &Wih[d * D_];
        const float* qb = &q[b * D_];
        for (int k = 0; k < D_; ++k) a += qb[k] * wi[k];
        u[d] = a;
        h[d] = 0.0f;
    }
    __syncthreads();
    for (int kk = 0; kk < 8; ++kk) {
        for (int d = tid; d < D_; d += 256) {
            float a = u[d];
            const float* wh = &Whh[d * D_];
            for (int k = 0; k < D_; ++k) a += h[k] * wh[k];
            hn[d] = tanhf(a);
        }
        __syncthreads();
        for (int d = tid; d < D_; d += 256) {
            h[d] = hn[d];
            H[(size_t)(b * 8 + kk) * D_ + d] = hn[d];
        }
        __syncthreads();
    }
}

// ---------------- masked attention readout ----------------
__global__ void __launch_bounds__(256)
attn_kernel(const float* __restrict__ H, const float* __restrict__ Vf,
            const int* __restrict__ lengths, float* __restrict__ out) {
    __shared__ float sc[8][S_];
    __shared__ float mx[8], sum[8];
    const int b = blockIdx.x, tid = threadIdx.x;
    const int len = lengths[b];
    for (int idx = tid; idx < 8 * S_; idx += 256) {
        int k = idx >> 7, s = idx & 127;
        float a = 0.0f;
        const float* hp = &H[(size_t)(b * 8 + k) * D_];
        const float* vp = &Vf[(size_t)(b * S_ + s) * DP];
        for (int d = 0; d < D_; ++d) a += hp[d] * vp[d];
        sc[k][s] = (s < len) ? a : -1e30f;     // exp -> exactly 0 weight
    }
    __syncthreads();
    if (tid < 8) {
        float m = -1e30f;
        for (int s = 0; s < S_; ++s) m = fmaxf(m, sc[tid][s]);
        float su = 0.0f;
        for (int s = 0; s < S_; ++s) su += __expf(sc[tid][s] - m);
        mx[tid] = m; sum[tid] = su;
    }
    __syncthreads();
    for (int idx = tid; idx < 8 * D_; idx += 256) {
        int k = idx / D_, d = idx % D_;
        float a = 0.0f;
        const float m = mx[k], inv = 1.0f / sum[k];
        for (int s = 0; s < S_; ++s)
            a += __expf(sc[k][s] - m) * Vf[(size_t)(b * S_ + s) * DP + d];
        out[(size_t)(b * 8 + k) * D_ + d] = a * inv;
    }
}

// ---------------- launcher ----------------
extern "C" void kernel_launch(void* const* d_in, const int* in_sizes, int n_in,
                              void* d_out, int out_size, void* d_ws, size_t ws_size,
                              hipStream_t stream) {
    const float* words = (const float*)d_in[0];
    const int*   lens  = (const int*)  d_in[1];
    const float* vocab = (const float*)d_in[2];
    const float* dft   = (const float*)d_in[3];
    const float* W     = (const float*)d_in[4];
    const float* lWih  = (const float*)d_in[5];
    const float* lWhh  = (const float*)d_in[6];
    const float* lbih  = (const float*)d_in[7];
    const float* lbhh  = (const float*)d_in[8];
    const float* rWih  = (const float*)d_in[9];
    const float* rWhh  = (const float*)d_in[10];
    const float* rbih  = (const float*)d_in[11];
    const float* rbhh  = (const float*)d_in[12];
    float* out = (float*)d_out;

    char* base = (char*)d_ws;
    size_t off = 0;
    auto take = [&](size_t bytes) {
        char* p = base + off;
        off = (off + bytes + 255) & ~(size_t)255;
        return p;
    };
    __bf16* vocab_b  = (__bf16*)take((size_t)VPAD * DP * 2);    // 12.8 MB row-major
    __bf16* vocabT_b = (__bf16*)take((size_t)DP * VPAD * 2);    // 12.8 MB transposed
    float*  Xf       = (float*) take((size_t)NROWS * DP * 4);   //  5.2 MB
    __bf16* Xb       = (__bf16*)take((size_t)NROWS * DP * 2);   //  2.6 MB
    float*  ldef     = (float*) take((size_t)NROWS * 4);
    float*  Vf       = (float*) take((size_t)NROWS * DP * 4);   //  5.2 MB
    float*  q        = (float*) take((size_t)B_ * D_ * 4);
    float*  H        = (float*) take((size_t)B_ * 8 * D_ * 4);
    (void)ws_size; (void)in_sizes; (void)n_in; (void)out_size;

    convert_vocab_kernel <<<(VPAD * DP + 255) / 256, 256, 0, stream>>>(vocab, vocab_b);
    convert_vocabT_kernel<<<(DP * VPAD + 255) / 256, 256, 0, stream>>>(vocab, vocabT_b);
    x_gemm_kernel<<<(NROWS * DP + 255) / 256, 256, 0, stream>>>(words, W, Xf, Xb);
    ldef_kernel<<<(NROWS + 255) / 256, 256, 0, stream>>>(Xf, dft, ldef);
    vocab_attend_kernel<<<NROWS / 16, 128, 0, stream>>>(vocab_b, vocabT_b, Xb, ldef, words, Vf);
    lstm_kernel<<<B_, 256, 0, stream>>>(Vf, lens, lWih, lWhh, lbih, lbhh, q);
    rnn_kernel<<<B_, 256, 0, stream>>>(q, rWih, rWhh, rbih, rbhh, H);
    attn_kernel<<<B_, 256, 0, stream>>>(H, Vf, lens, out);
}
